// EfficientAttention3D_28398323761333
// MI455X (gfx1250) — matrix operand order repaired and checked by reading, not compiled, not run on hardware
//
#include <hip/hip_runtime.h>
#include <stdint.h>
#include <stddef.h>

#define IN_C   128
#define HEADS  8
#define NCLS   4
#define DL     65536            // L = 16*64*64
#define NB     2
#define NCOLS  128              // columns per block

typedef __attribute__((ext_vector_type(16))) __bf16 v16bf;
typedef __attribute__((ext_vector_type(8)))  float  v8f;

union V16BF { uint32_t u[8]; v16bf v; };

__device__ __forceinline__ uint16_t f2bf(float f) {
  uint32_t u = __float_as_uint(f);
  uint32_t r = u + 0x7FFFu + ((u >> 16) & 1u);
  return (uint16_t)(r >> 16);
}
__device__ __forceinline__ uint32_t pack_bf2(float lo, float hi) {
  return (uint32_t)f2bf(lo) | ((uint32_t)f2bf(hi) << 16);
}

// ---- CDNA5 async global->LDS copy (ASYNCcnt path), 16B per lane ----
__device__ __forceinline__ void async_b128(uint32_t lds_byte_off, const void* g) {
  asm volatile("global_load_async_to_lds_b128 %0, %1, off"
               :: "v"(lds_byte_off), "v"((unsigned long long)(uintptr_t)g)
               : "memory");
}
__device__ __forceinline__ void wait_async0() {
  asm volatile("s_wait_asynccnt 0" ::: "memory");
}
__device__ __forceinline__ uint32_t lds_off(const void* p) {
  return (uint32_t)(uintptr_t)p;   // LDS aperture keeps offset in addr[31:0]
}

// ---------------- workspace layout (bytes) ----------------
static constexpr size_t WS_QSM = 0;                                   // bf16 [2][128][DL]
static constexpr size_t WS_WQ  = WS_QSM + (size_t)NB * 128 * DL * 2;  // bf16 [128][128]
static constexpr size_t WS_WV  = WS_WQ + 128 * 128 * 2;
static constexpr size_t WS_WK  = WS_WV + 128 * 128 * 2;
static constexpr size_t WS_M2  = WS_WK + 128 * 128 * 2;               // bf16 [2][128][128]
static constexpr size_t WS_CTX = WS_M2 + (size_t)NB * 128 * 128 * 2;  // f32  [2][8][16][16]
static constexpr size_t WS_SUM = WS_CTX + (size_t)NB * 8 * 16 * 16 * 4; // f32 [2][128]

// ---------------- output layout (floats) ----------------
static constexpr size_t OUT_CTX = (size_t)NB * 128 * DL;   // context_last [2][16][16]
static constexpr size_t OUT_CLS = OUT_CTX + NB * 16 * 16;  // CLS_out      [2][16][4]

// 4 bf16 K-steps of 32 against 8 16-row A tiles, f32 accumulate
__device__ __forceinline__ void gemm_tiles(const uint16_t* __restrict__ W,
                                           const V16BF Breg[4], int mIdx, int half,
                                           v8f acc8[8]) {
#pragma unroll
  for (int t = 0; t < 8; ++t) {
    v8f acc = {0.f, 0.f, 0.f, 0.f, 0.f, 0.f, 0.f, 0.f};
#pragma unroll
    for (int ks = 0; ks < 4; ++ks) {
      V16BF a;
#pragma unroll
      for (int r = 0; r < 8; ++r) {
        int kb = (r << 1) + (half << 4) + (ks << 5);
        a.u[r] = *(const uint32_t*)&W[(t * 16 + mIdx) * 128 + kb];
      }
      acc = __builtin_amdgcn_wmma_f32_16x16x32_bf16(false, a.v, false, Breg[ks].v,
                                                    (short)0, acc, false, false);
    }
    acc8[t] = acc;
  }
}

// -------- kernel 1: weight bf16 conversion + CLS contributions (init ctx/sumexp) --------
__global__ void ea3d_init(const float* __restrict__ CLS,
                          const float* __restrict__ Wq, const float* __restrict__ Wv,
                          const float* __restrict__ Wk,
                          uint16_t* __restrict__ wq16, uint16_t* __restrict__ wv16,
                          uint16_t* __restrict__ wk16,
                          float* __restrict__ ctxAcc, float* __restrict__ sumexp) {
  const int tid = threadIdx.x;
  for (int i = tid; i < 128 * 128; i += 256) {
    wq16[i] = f2bf(Wq[i]);
    wv16[i] = f2bf(Wv[i]);
    wk16[i] = f2bf(Wk[i]);
  }
  for (int i = tid; i < NB * 128; i += 256) {
    int n = i >> 7, c = i & 127;
    const float* p = CLS + (n * 128 + c) * NCLS;
    float s = 0.f;
    for (int j = 0; j < NCLS; ++j) s += __expf(p[j]);
    sumexp[i] = s;   // fresh init each launch (pass1 atomically adds on top)
  }
  for (int i = tid; i < NB * 8 * 16 * 16; i += 256) {
    int v = i & 15, k = (i >> 4) & 15, h = (i >> 8) & 7, n = i >> 11;
    const float* pk = CLS + (n * 128 + h * 16 + k) * NCLS;
    const float* pv = CLS + (n * 128 + h * 16 + v) * NCLS;
    float s = 0.f;
    for (int j = 0; j < NCLS; ++j) s += __expf(pk[j]) * pv[j];
    ctxAcc[i] = s;
  }
}

// -------- kernel 2: fused K/Q/V GEMMs + q-softmax + exp(keys) + context partials --------
__global__ __launch_bounds__(256)
void ea3d_pass1(const float* __restrict__ x,
                const uint16_t* __restrict__ wq16, const uint16_t* __restrict__ wv16,
                const uint16_t* __restrict__ wk16,
                const float* __restrict__ bq, const float* __restrict__ bv,
                const float* __restrict__ bk,
                uint16_t* __restrict__ qsm, float* __restrict__ ctxAcc,
                float* __restrict__ sumexp) {
  __shared__ float    sXf[128 * 128];   // 64KB: x tile f32 [c][l]; later reused as E (bf16 [k][l])
  __shared__ uint16_t sV[128 * 128];    // 32KB: V tile bf16 [v][l]
  uint16_t* sE = (uint16_t*)sXf;        // alias (32KB of the 64KB region)

  const int n    = blockIdx.y;
  const int l0   = blockIdx.x * NCOLS;
  const int tid  = threadIdx.x;
  const int wave = tid >> 5;
  const int lane = tid & 31;
  const int half = lane >> 4;
  const int nIdx = lane & 15;
  const int c0   = wave * 16;           // this wave's 16 columns

  // --- async-stage x tile (f32, [c][l]) straight into LDS: 4096 chunks of 16B ---
  {
    const float* xsrc = x + (size_t)(n * IN_C) * DL + l0;
    const uint32_t base = lds_off(sXf);
    for (int i = tid; i < 4096; i += 256) {
      const char* g = (const char*)(xsrc + (size_t)(i >> 5) * DL) + (i & 31) * 16;
      async_b128(base + i * 16, g);
    }
    wait_async0();
  }
  __syncthreads();

  // B operands (x tile) for all 4 K-steps: f32 pair -> packed bf16, reused for 3 GEMMs
  V16BF Breg[4];
#pragma unroll
  for (int ks = 0; ks < 4; ++ks)
#pragma unroll
    for (int r = 0; r < 8; ++r) {
      int c = ks * 32 + r * 2 + half * 16;
      float lo = sXf[c * 128 + c0 + nIdx];
      float hi = sXf[(c + 1) * 128 + c0 + nIdx];
      Breg[ks].u[r] = pack_bf2(lo, hi);
    }
  __syncthreads();   // all waves done reading x before sXf region is reused for E

  v8f acc8[8];

  // ---- queries: GEMM + bias + per-head softmax over 16 channels -> qsm (bf16) ----
  gemm_tiles(wq16, Breg, nIdx, half, acc8);
#pragma unroll
  for (int t = 0; t < 8; ++t) {
    v8f acc = acc8[t];
#pragma unroll
    for (int r = 0; r < 8; ++r) acc[r] += bq[t * 16 + r + half * 8];
    float m = acc[0];
#pragma unroll
    for (int r = 1; r < 8; ++r) m = fmaxf(m, acc[r]);
    m = fmaxf(m, __shfl_xor(m, 16, 32));
    float s = 0.f;
#pragma unroll
    for (int r = 0; r < 8; ++r) { acc[r] = __expf(acc[r] - m); s += acc[r]; }
    s += __shfl_xor(s, 16, 32);
    float inv = 1.0f / s;
#pragma unroll
    for (int r = 0; r < 8; ++r) {
      int row = t * 16 + r + half * 8;
      qsm[((size_t)(n * 128 + row)) * DL + l0 + c0 + nIdx] = f2bf(acc[r] * inv);
    }
  }

  // ---- values: GEMM + bias -> LDS V tile [v][l] (bf16) ----
  gemm_tiles(wv16, Breg, nIdx, half, acc8);
#pragma unroll
  for (int t = 0; t < 8; ++t)
#pragma unroll
    for (int r = 0; r < 8; ++r) {
      int row = t * 16 + r + half * 8;
      sV[row * 128 + c0 + nIdx] = f2bf(acc8[t][r] + bv[row]);
    }

  // ---- keys: GEMM + bias, E = exp(keys); row sums -> global sumexp ----
  gemm_tiles(wk16, Breg, nIdx, half, acc8);
#pragma unroll
  for (int t = 0; t < 8; ++t)
#pragma unroll
    for (int r = 0; r < 8; ++r) {
      int row = t * 16 + r + half * 8;
      acc8[t][r] = __expf(acc8[t][r] + bk[row]);
    }
#pragma unroll
  for (int t = 0; t < 8; ++t)
#pragma unroll
    for (int r = 0; r < 8; ++r) {
      float s = acc8[t][r];
      s += __shfl_xor(s, 1, 32);
      s += __shfl_xor(s, 2, 32);
      s += __shfl_xor(s, 4, 32);
      s += __shfl_xor(s, 8, 32);
      if (nIdx == 0) {
        int row = t * 16 + r + half * 8;
        atomicAdd(&sumexp[n * 128 + row], s);
      }
    }

  // write E into sE (x tile region; all waves finished reading it at the barrier above)
#pragma unroll
  for (int t = 0; t < 8; ++t)
#pragma unroll
    for (int r = 0; r < 8; ++r) {
      int row = t * 16 + r + half * 8;
      sE[row * 128 + c0 + nIdx] = f2bf(acc8[t][r]);
    }
  __syncthreads();   // V writes + E writes visible before cross-wave reads

  // ---- context partial: one wave per head, ctx_h += E_h @ V_h^T over 128 cols ----
  const int h = wave;
  v8f ctx = {0.f, 0.f, 0.f, 0.f, 0.f, 0.f, 0.f, 0.f};
#pragma unroll
  for (int ks = 0; ks < 4; ++ks) {
    V16BF a, b;
#pragma unroll
    for (int r = 0; r < 8; ++r) {
      int kb = (r << 1) + (half << 4) + (ks << 5);
      a.u[r] = *(const uint32_t*)&sE[(h * 16 + nIdx) * 128 + kb];
      int kb2 = ks * 32 + r * 2 + half * 16;
      b.u[r] = *(const uint32_t*)&sV[(h * 16 + nIdx) * 128 + kb2];
    }
    ctx = __builtin_amdgcn_wmma_f32_16x16x32_bf16(false, a.v, false, b.v,
                                                  (short)0, ctx, false, false);
  }
#pragma unroll
  for (int r = 0; r < 8; ++r) {
    int k = r + half * 8;
    atomicAdd(&ctxAcc[((n * 8 + h) * 16 + k) * 16 + nIdx], ctx[r]);
  }
}

// -------- kernel 3: normalize context, emit context_last & CLS_out, build M2 --------
__global__ void ea3d_finalize(const float* __restrict__ Wr,
                              const float* __restrict__ CLS,
                              const float* __restrict__ ctxAcc,
                              const float* __restrict__ sumexp,
                              uint16_t* __restrict__ m2, float* __restrict__ out) {
  __shared__ float ctxs[8 * 16 * 16];
  const int n = blockIdx.x;
  const int tid = threadIdx.x;
  for (int i = tid; i < 2048; i += 256) {
    int v = i & 15, k = (i >> 4) & 15, h = i >> 8;
    float cf = ctxAcc[n * 2048 + i] / sumexp[n * 128 + h * 16 + k];
    ctxs[i] = cf;
    if (h == 7) out[OUT_CTX + n * 256 + k * 16 + v] = cf;
  }
  __syncthreads();
  // M2[o][(h,k)] = sum_v Wr[o][16h+v] * ctx_h[k][v]
  for (int i = tid; i < 128 * 128; i += 256) {
    int o = i >> 7, c = i & 127;
    int h = c >> 4, k = c & 15;
    float s = 0.f;
    for (int v = 0; v < 16; ++v) s += Wr[o * 128 + h * 16 + v] * ctxs[(h * 16 + k) * 16 + v];
    m2[(size_t)n * 128 * 128 + i] = f2bf(s);
  }
  // CLS_out[v][j] = sum_k ctx_7[k][v] * softmax_k(CLS[112+k][j])
  if (tid < 64) {
    int v = tid >> 2, j = tid & 3;
    float mq = -1e30f;
    for (int k = 0; k < 16; ++k) mq = fmaxf(mq, CLS[(n * 128 + 112 + k) * NCLS + j]);
    float ssum = 0.f, e[16];
    for (int k = 0; k < 16; ++k) {
      e[k] = __expf(CLS[(n * 128 + 112 + k) * NCLS + j] - mq);
      ssum += e[k];
    }
    float r = 0.f;
    for (int k = 0; k < 16; ++k) r += ctxs[(7 * 16 + k) * 16 + v] * (e[k] / ssum);
    out[OUT_CLS + n * 64 + v * 4 + j] = r;
  }
}

// -------- kernel 4: attention = M2 @ q_sm + br  (bf16 WMMA GEMM, async-staged) --------
__global__ __launch_bounds__(256)
void ea3d_pass2(const uint16_t* __restrict__ qsm, const uint16_t* __restrict__ m2,
                const float* __restrict__ br, float* __restrict__ out) {
  __shared__ uint16_t sQ[128 * 128];   // q tile bf16 [c][l] (natural order, async-copied)
  __shared__ uint16_t sW[128 * 128];   // M2 bf16 [o][c]     (async-copied)
  const int n    = blockIdx.y;
  const int l0   = blockIdx.x * NCOLS;
  const int tid  = threadIdx.x;
  const int wave = tid >> 5;
  const int lane = tid & 31;
  const int half = lane >> 4;
  const int nIdx = lane & 15;
  const int c0   = wave * 16;

  {
    const uint16_t* qsrc = qsm + (size_t)(n * 128) * DL + l0;
    const char*     wsrc = (const char*)(m2 + (size_t)n * 16384);
    const uint32_t qbase = lds_off(sQ);
    const uint32_t wbase = lds_off(sW);
    for (int i = tid; i < 2048; i += 256) {   // 2048 x 16B = 32KB each
      const char* g = (const char*)(qsrc + (size_t)(i >> 4) * DL) + (i & 15) * 16;
      async_b128(qbase + i * 16, g);
      async_b128(wbase + i * 16, wsrc + i * 16);
    }
    wait_async0();
  }
  __syncthreads();

  V16BF Breg[4];
#pragma unroll
  for (int ks = 0; ks < 4; ++ks)
#pragma unroll
    for (int r = 0; r < 8; ++r) {
      int c = ks * 32 + r * 2 + half * 16;
      uint32_t lo = sQ[c * 128 + c0 + nIdx];
      uint32_t hi = sQ[(c + 1) * 128 + c0 + nIdx];
      Breg[ks].u[r] = lo | (hi << 16);
    }

  v8f acc8[8];
  gemm_tiles(sW, Breg, nIdx, half, acc8);
#pragma unroll
  for (int t = 0; t < 8; ++t)
#pragma unroll
    for (int r = 0; r < 8; ++r) {
      int row = t * 16 + r + half * 8;
      out[((size_t)(n * 128 + row)) * DL + l0 + c0 + nIdx] = acc8[t][r] + br[row];
    }
}

extern "C" void kernel_launch(void* const* d_in, const int* in_sizes, int n_in,
                              void* d_out, int out_size, void* d_ws, size_t ws_size,
                              hipStream_t stream) {
  (void)in_sizes; (void)n_in; (void)out_size; (void)ws_size;
  const float* x   = (const float*)d_in[0];
  const float* CLS = (const float*)d_in[1];
  const float* Wk  = (const float*)d_in[2];
  const float* bk  = (const float*)d_in[3];
  const float* Wq  = (const float*)d_in[4];
  const float* bq  = (const float*)d_in[5];
  const float* Wv  = (const float*)d_in[6];
  const float* bv  = (const float*)d_in[7];
  const float* Wr  = (const float*)d_in[8];
  const float* br  = (const float*)d_in[9];
  float* out = (float*)d_out;

  char* ws = (char*)d_ws;
  uint16_t* qsm    = (uint16_t*)(ws + WS_QSM);
  uint16_t* wq16   = (uint16_t*)(ws + WS_WQ);
  uint16_t* wv16   = (uint16_t*)(ws + WS_WV);
  uint16_t* wk16   = (uint16_t*)(ws + WS_WK);
  uint16_t* m2     = (uint16_t*)(ws + WS_M2);
  float*    ctxAcc = (float*)(ws + WS_CTX);
  float*    sumexp = (float*)(ws + WS_SUM);

  ea3d_init<<<1, 256, 0, stream>>>(CLS, Wq, Wv, Wk, wq16, wv16, wk16, ctxAcc, sumexp);
  ea3d_pass1<<<dim3(DL / NCOLS, NB), 256, 0, stream>>>(x, wq16, wv16, wk16, bq, bv, bk,
                                                       qsm, ctxAcc, sumexp);
  ea3d_finalize<<<NB, 256, 0, stream>>>(Wr, CLS, ctxAcc, sumexp, m2, out);
  ea3d_pass2<<<dim3(DL / NCOLS, NB), 256, 0, stream>>>(qsm, m2, br, out);
}
